// TriangleMultiplication_60017872994855
// MI455X (gfx1250) — matrix operand order repaired and checked by reading, not compiled, not run on hardware
//
#include <hip/hip_runtime.h>

// ---------------------------------------------------------------------------
// Triangle multiplication (outgoing) for MI455X / gfx1250.
//   N = 512 (seq), D = H = 128.  All matmuls: bf16 x bf16 -> f32 WMMA
//   (v_wmma_f32_16x16x32_bf16), LayerNorms / sigmoid in f32.
// Workspace layout (requires 256 MiB):
//   leftT  [128][512][512] bf16   (d-major:  leftT[d][i][k])
//   rightT [128][512][512] bf16
//   gateT  [512*512][128]  bf16   (position-major, d contiguous)
//   outT   [128][512][512] bf16   (pre-LN triangle output, d-major)
// ---------------------------------------------------------------------------

#define N_SEQ 512
#define NN (N_SEQ * N_SEQ) // 262144
#define EPS_LN 1e-5f
#define K3_BUFSZ (2 * 128 * 40) // one double-buffer slot: A panel + B panel

typedef __attribute__((ext_vector_type(8))) float v8f;
typedef __attribute__((ext_vector_type(8))) __bf16 v8bf;
typedef __attribute__((ext_vector_type(16))) __bf16 v16bf;

__device__ __forceinline__ unsigned short f2bf(float f) {
  return __builtin_bit_cast(unsigned short, (__bf16)f); // native v_cvt, RTNE
}
__device__ __forceinline__ float bf2f(unsigned short h) {
  unsigned u = ((unsigned)h) << 16;
  return __builtin_bit_cast(float, u);
}
// Fast sigmoid: v_exp_f32 + v_rcp_f32 (avoid the IEEE div fixup chain).
__device__ __forceinline__ float sigmoidf(float x) {
  return __builtin_amdgcn_rcpf(1.0f + __expf(-x));
}

// CDNA5 async global->LDS copy (GLOBAL_LOAD_ASYNC_TO_LDS_B128, ASYNCcnt).
// LDS byte address = low 32 bits of the generic __shared__ pointer.
__device__ __forceinline__ void async_copy_b128(unsigned short* lds_dst,
                                                const unsigned short* gsrc) {
  const unsigned loff = (unsigned)(size_t)lds_dst;
  asm volatile("global_load_async_to_lds_b128 %0, %1, off"
               :: "v"(loff), "v"(gsrc)
               : "memory");
}
#define WAIT_ASYNCCNT(n) asm volatile("s_wait_asynccnt " #n ::: "memory")

// A-fragment (16x32 bf16, MxK) per CDNA5 ISA 7.12.2:
//  lanes 0-15 : M=lane,  K = k0+0..7  then k0+16..23
//  lanes 16-31: M=lane-16, K = k0+8..15 then k0+24..31
__device__ __forceinline__ v16bf load_a_frag(const unsigned short* base, int pitch,
                                             int m0, int k0, int lane) {
  const int l = lane & 15;
  const int koff = k0 + ((lane & 16) ? 8 : 0);
  const unsigned short* p = base + (m0 + l) * pitch + koff;
  v8bf lo = *(const v8bf*)(p);
  v8bf hi = *(const v8bf*)(p + 16);
  return __builtin_shufflevector(lo, hi, 0, 1, 2, 3, 4, 5, 6, 7, 8, 9, 10, 11, 12, 13, 14, 15);
}

// B-fragment (32x16 bf16, KxN): lanes 0-15: N=lane, K=k0+0..7 then k0+16..23;
//                               lanes 16-31: N=lane-16, K=k0+8..15 then k0+24..31.
// LDS holds B row-major as [n][k] (weight rows / rhs rows contiguous in K).
__device__ __forceinline__ v16bf load_b_frag(const unsigned short* base, int pitch,
                                             int n0, int k0, int lane) {
  const int l = lane & 15;
  const int koff = k0 + ((lane & 16) ? 8 : 0);
  const unsigned short* p = base + (n0 + l) * pitch + koff;
  v8bf lo = *(const v8bf*)(p);
  v8bf hi = *(const v8bf*)(p + 16);
  return __builtin_shufflevector(lo, hi, 0, 1, 2, 3, 4, 5, 6, 7, 8, 9, 10, 11, 12, 13, 14, 15);
}

__device__ __forceinline__ v8f wmma_bf16(v16bf a, v16bf b, v8f c) {
  return __builtin_amdgcn_wmma_f32_16x16x32_bf16(false, a, false, b, (short)0, c, false, false);
}

// ---------------------------------------------------------------------------
// Kernel 1: LN(pair) -> fused projections (p_in|g_in|g_out = 640 cols) ->
//           gating -> transposed bf16 stores of left/right (d-major) + gate.
// One block = 128 consecutive pair rows (one i, 128 consecutive j).
// LDS: W 640x136 bf16 (170KiB) + A 128x136 (34KiB) + 2 stage bufs (68KiB).
// ---------------------------------------------------------------------------
__global__ __launch_bounds__(256) void k1_fused_in(
    const float* __restrict__ pair, const float* __restrict__ ln_w,
    const float* __restrict__ ln_b, const float* __restrict__ p_in_w,
    const float* __restrict__ g_in_w, const float* __restrict__ g_out_w,
    unsigned short* __restrict__ leftT, unsigned short* __restrict__ rightT,
    unsigned short* __restrict__ gateT) {
  extern __shared__ unsigned short smem1[];
  unsigned short* wlds = smem1;              // [640][136]
  unsigned short* alds = wlds + 640 * 136;   // [128][136]
  unsigned short* stgL = alds + 128 * 136;   // [128 h][136 m]
  unsigned short* stgR = stgL + 128 * 136;

  const int tid = threadIdx.x;
  const int lane = tid & 31;
  const int wv = tid >> 5;
  const int R0 = blockIdx.x * 128;
  const int i_row = R0 >> 9;
  const int j0 = R0 & (N_SEQ - 1);

  // Combined weights -> LDS bf16, rows = output column index (0..639).
  for (int idx = tid; idx < 640 * 32; idx += 256) {
    const int row = idx >> 5, c4 = (idx & 31) << 2;
    const float* src = (row < 256) ? (p_in_w + row * 128)
                     : (row < 512) ? (g_in_w + (row - 256) * 128)
                                   : (g_out_w + (row - 512) * 128);
    const float4 v = *(const float4*)(src + c4);
    unsigned short* dst = wlds + row * 136 + c4;
    dst[0] = f2bf(v.x); dst[1] = f2bf(v.y); dst[2] = f2bf(v.z); dst[3] = f2bf(v.w);
  }

  // LayerNorm: one wave per row, 4 channels per lane, shfl_xor reduction.
  const float4 lw = *(const float4*)(ln_w + 4 * lane);
  const float4 lb = *(const float4*)(ln_b + 4 * lane);
  for (int r = wv; r < 128; r += 8) {
    const float4 x = *(const float4*)(pair + (size_t)(R0 + r) * 128 + 4 * lane);
    float s = x.x + x.y + x.z + x.w;
    float s2 = x.x * x.x + x.y * x.y + x.z * x.z + x.w * x.w;
    for (int o = 16; o > 0; o >>= 1) {
      s += __shfl_xor(s, o);
      s2 += __shfl_xor(s2, o);
    }
    const float mu = s * (1.0f / 128.0f);
    const float rstd = rsqrtf(s2 * (1.0f / 128.0f) - mu * mu + EPS_LN);
    unsigned short* dst = alds + r * 136 + 4 * lane;
    dst[0] = f2bf((x.x - mu) * rstd * lw.x + lb.x);
    dst[1] = f2bf((x.y - mu) * rstd * lw.y + lb.y);
    dst[2] = f2bf((x.z - mu) * rstd * lw.z + lb.z);
    dst[3] = f2bf((x.w - mu) * rstd * lw.w + lb.w);
  }
  __syncthreads();

  // left = sig(g[:,0:128]) * p[:,0:128]; right = sig(g[:,128:256]) * p[:,128:256] / L
  // Column-tile pairs (t, t+16): p cols 16t, g cols 256+16t.  t<8 -> left, t>=8 -> right.
  for (int t = wv; t < 16; t += 8) {
    v16bf bp[4], bg[4];
#pragma unroll
    for (int ks = 0; ks < 4; ++ks) {
      bp[ks] = load_b_frag(wlds, 136, 16 * t, 32 * ks, lane);
      bg[ks] = load_b_frag(wlds, 136, 256 + 16 * t, 32 * ks, lane);
    }
    const bool isR = (t >= 8);
    unsigned short* stg = isR ? stgR : stgL;
    const float scale = isR ? (1.0f / 512.0f) : 1.0f;
    const int hbase = (t & 7) * 16;
    for (int mt = 0; mt < 8; ++mt) {
      v8f ap = {}, ag = {};
#pragma unroll
      for (int ks = 0; ks < 4; ++ks) {
        const v16bf a = load_a_frag(alds, 136, 16 * mt, 32 * ks, lane);
        ap = wmma_bf16(a, bp[ks], ap);
        ag = wmma_bf16(a, bg[ks], ag);
      }
      const int n = lane & 15;
      const int mr = 16 * mt + ((lane & 16) ? 8 : 0);
#pragma unroll
      for (int r = 0; r < 8; ++r) {
        const float v = ap[r] * sigmoidf(ag[r]) * scale;
        stg[(hbase + n) * 136 + (mr + r)] = f2bf(v); // transposed stage [h][m]
      }
    }
  }

  // gate = sigmoid(pair_norm @ g_out_w^T): cols 512..639, stored [pos][d].
  {
    const int t = 32 + wv;
    v16bf bgt[4];
#pragma unroll
    for (int ks = 0; ks < 4; ++ks)
      bgt[ks] = load_b_frag(wlds, 136, 16 * t, 32 * ks, lane);
    for (int mt = 0; mt < 8; ++mt) {
      v8f ac = {};
#pragma unroll
      for (int ks = 0; ks < 4; ++ks) {
        const v16bf a = load_a_frag(alds, 136, 16 * mt, 32 * ks, lane);
        ac = wmma_bf16(a, bgt[ks], ac);
      }
      const int dn = 16 * wv + (lane & 15);
      const int mr = 16 * mt + ((lane & 16) ? 8 : 0);
#pragma unroll
      for (int r = 0; r < 8; ++r)
        gateT[(size_t)(R0 + mr + r) * 128 + dn] = f2bf(sigmoidf(ac[r]));
    }
  }
  __syncthreads();

  // Cooperative transposed writeout: leftT[d][i][j0+m], 256B runs per (d,i).
  for (int c = tid; c < 2048; c += 256) {
    const int h = c >> 4, mo = (c & 15) << 3;
    const uint4 vl = *(const uint4*)(stgL + h * 136 + mo);
    const uint4 vr = *(const uint4*)(stgR + h * 136 + mo);
    const size_t g = ((size_t)h * N_SEQ + i_row) * N_SEQ + (j0 + mo);
    *(uint4*)(leftT + g) = vl;
    *(uint4*)(rightT + g) = vr;
  }
}

// ---------------------------------------------------------------------------
// Kernel 3: per-d triangle GEMM  out[d][i][j] = sum_k left[d][i][k]*right[d][j][k]
// Block = one (d, 128x128 tile). Wave w owns M-strip 16w, all 8 N-tiles.
// Panels double-buffered in LDS via GLOBAL_LOAD_ASYNC_TO_LDS_B128: panel k+1
// streams in (ASYNCcnt) while WMMAs consume panel k.
// ---------------------------------------------------------------------------
__global__ __launch_bounds__(256) void k3_triangle(
    const unsigned short* __restrict__ leftT,
    const unsigned short* __restrict__ rightT,
    unsigned short* __restrict__ outT) {
  extern __shared__ unsigned short smem3[];
  // double buffer selected by integer offset (no LDS pointer arrays!)

  const int tid = threadIdx.x, lane = tid & 31, wv = tid >> 5;
  const int d = blockIdx.x >> 4;
  const int i0 = ((blockIdx.x >> 2) & 3) * 128;
  const int j0 = (blockIdx.x & 3) * 128;
  const unsigned short* A = leftT + (size_t)d * NN;
  const unsigned short* B = rightT + (size_t)d * NN;

  // Per-thread panel chunks: 2 x 16B from A rows and 2 x 16B from B rows.
  const int r0c = tid >> 2, co0 = (tid & 3) << 3;        // chunk tid
  const int r1c = (tid + 256) >> 2, co1 = co0;           // chunk tid+256

  v8f acc[8];
#pragma unroll
  for (int t = 0; t < 8; ++t) acc[t] = (v8f){};

  // Prologue: issue panel 0 into buffer 0 (4 async b128 per thread / per wave).
  {
    unsigned short* aP = smem3;
    unsigned short* bP = smem3 + 128 * 40;
    async_copy_b128(aP + r0c * 40 + co0, A + (size_t)(i0 + r0c) * N_SEQ + co0);
    async_copy_b128(aP + r1c * 40 + co1, A + (size_t)(i0 + r1c) * N_SEQ + co1);
    async_copy_b128(bP + r0c * 40 + co0, B + (size_t)(j0 + r0c) * N_SEQ + co0);
    async_copy_b128(bP + r1c * 40 + co1, B + (size_t)(j0 + r1c) * N_SEQ + co1);
  }

  for (int ks = 0; ks < 16; ++ks) {
    const int cur = ks & 1;
    if (ks + 1 < 16) { // stream next panel into the other buffer
      const int k0 = 32 * (ks + 1);
      unsigned short* aP = smem3 + (1 - cur) * K3_BUFSZ;
      unsigned short* bP = aP + 128 * 40;
      async_copy_b128(aP + r0c * 40 + co0, A + (size_t)(i0 + r0c) * N_SEQ + k0 + co0);
      async_copy_b128(aP + r1c * 40 + co1, A + (size_t)(i0 + r1c) * N_SEQ + k0 + co1);
      async_copy_b128(bP + r0c * 40 + co0, B + (size_t)(j0 + r0c) * N_SEQ + k0 + co0);
      async_copy_b128(bP + r1c * 40 + co1, B + (size_t)(j0 + r1c) * N_SEQ + k0 + co1);
      WAIT_ASYNCCNT(4); // oldest 4 (panel ks) complete; panel ks+1 in flight
    } else {
      WAIT_ASYNCCNT(0);
    }
    __syncthreads();

    const unsigned short* aP = smem3 + cur * K3_BUFSZ;
    const unsigned short* bP = aP + 128 * 40;
    const v16bf af = load_a_frag(aP, 40, 16 * wv, 0, lane);
#pragma unroll
    for (int nt = 0; nt < 8; ++nt) {
      const v16bf bfrag = load_b_frag(bP, 40, 16 * nt, 0, lane);
      acc[nt] = wmma_bf16(af, bfrag, acc[nt]);
    }
    __syncthreads(); // all waves done with this buffer before it is refilled
  }

  const int n = lane & 15;
  const int mr = 16 * wv + ((lane & 16) ? 8 : 0);
#pragma unroll
  for (int nt = 0; nt < 8; ++nt)
#pragma unroll
    for (int r = 0; r < 8; ++r)
      outT[(size_t)d * NN + (size_t)(i0 + mr + r) * N_SEQ + (j0 + 16 * nt + n)] =
          f2bf(acc[nt][r]);
}

// ---------------------------------------------------------------------------
// Kernel 4: gather 128 d per 16x16 position tile, LN over d, project by
// p_out_w (WMMA), multiply by gate, store f32 output.
// ---------------------------------------------------------------------------
__global__ __launch_bounds__(256) void k4_out(
    const unsigned short* __restrict__ outT,
    const unsigned short* __restrict__ gateT, const float* __restrict__ lnw,
    const float* __restrict__ lnb, const float* __restrict__ p_out_w,
    float* __restrict__ out) {
  extern __shared__ unsigned char smem4[];
  float* S = (float*)smem4;                                // [256 pos][132] f32
  unsigned short* A = (unsigned short*)(S + 256 * 132);    // [256][136] bf16
  unsigned short* W = A + 256 * 136;                       // [128][136] bf16

  const int tid = threadIdx.x, lane = tid & 31, wv = tid >> 5;
  const int i0 = (blockIdx.x >> 5) << 4;
  const int j0 = (blockIdx.x & 31) << 4;
  const int pi = tid >> 4, pj = tid & 15;

  // Gather out_t[d][i][j] over all d for this 16x16 tile (coalesced over j).
  for (int d0 = 0; d0 < 128; ++d0)
    S[tid * 132 + d0] = bf2f(outT[(size_t)d0 * NN + (size_t)(i0 + pi) * N_SEQ + (j0 + pj)]);

  // p_out_w [128 out][128 in] -> LDS bf16.
  for (int idx = tid; idx < 128 * 32; idx += 256) {
    const int row = idx >> 5, c4 = (idx & 31) << 2;
    const float4 v = *(const float4*)(p_out_w + row * 128 + c4);
    unsigned short* dst = W + row * 136 + c4;
    dst[0] = f2bf(v.x); dst[1] = f2bf(v.y); dst[2] = f2bf(v.z); dst[3] = f2bf(v.w);
  }
  __syncthreads();

  // LayerNorm over d, one wave per row.
  const float4 lw = *(const float4*)(lnw + 4 * lane);
  const float4 lb = *(const float4*)(lnb + 4 * lane);
  for (int row = wv; row < 256; row += 8) {
    const float4 x = *(const float4*)(S + row * 132 + 4 * lane);
    float s = x.x + x.y + x.z + x.w;
    float s2 = x.x * x.x + x.y * x.y + x.z * x.z + x.w * x.w;
    for (int o = 16; o > 0; o >>= 1) {
      s += __shfl_xor(s, o);
      s2 += __shfl_xor(s2, o);
    }
    const float mu = s * (1.0f / 128.0f);
    const float rstd = rsqrtf(s2 * (1.0f / 128.0f) - mu * mu + EPS_LN);
    unsigned short* dst = A + row * 136 + 4 * lane;
    dst[0] = f2bf((x.x - mu) * rstd * lw.x + lb.x);
    dst[1] = f2bf((x.y - mu) * rstd * lw.y + lb.y);
    dst[2] = f2bf((x.z - mu) * rstd * lw.z + lb.z);
    dst[3] = f2bf((x.w - mu) * rstd * lw.w + lb.w);
  }
  __syncthreads();

  // Projection GEMM (M=256 pos, N=128 d_out, K=128) + gate epilogue.
  for (int mt = 2 * wv; mt < 2 * wv + 2; ++mt) {
    v16bf af[4];
#pragma unroll
    for (int ks = 0; ks < 4; ++ks) af[ks] = load_a_frag(A, 136, 16 * mt, 32 * ks, lane);
#pragma unroll
    for (int nt = 0; nt < 8; ++nt) {
      v8f ac = {};
#pragma unroll
      for (int ks = 0; ks < 4; ++ks)
        ac = wmma_bf16(af[ks], load_b_frag(W, 136, 16 * nt, 32 * ks, lane), ac);
      const int n = 16 * nt + (lane & 15);
      const int mr = 16 * mt + ((lane & 16) ? 8 : 0);
#pragma unroll
      for (int r = 0; r < 8; ++r) {
        const int pos = mr + r;
        const size_t fp = (size_t)(i0 + (pos >> 4)) * N_SEQ + (j0 + (pos & 15));
        const float g = bf2f(gateT[fp * 128 + n]);
        out[fp * 128 + n] = ac[r] * g;
      }
    }
  }
}

// ---------------------------------------------------------------------------
extern "C" void kernel_launch(void* const* d_in, const int* in_sizes, int n_in,
                              void* d_out, int out_size, void* d_ws,
                              size_t ws_size, hipStream_t stream) {
  const float* pair = (const float*)d_in[0];
  const float* ln_in_w = (const float*)d_in[1];
  const float* ln_in_b = (const float*)d_in[2];
  const float* p_in_w = (const float*)d_in[3];
  const float* g_in_w = (const float*)d_in[4];
  const float* ln_out_w = (const float*)d_in[5];
  const float* ln_out_b = (const float*)d_in[6];
  const float* p_out_w = (const float*)d_in[7];
  const float* g_out_w = (const float*)d_in[8];

  unsigned short* leftT = (unsigned short*)d_ws;          // 64 MiB each
  unsigned short* rightT = leftT + (size_t)128 * NN;
  unsigned short* gateT = rightT + (size_t)128 * NN;
  unsigned short* outT = gateT + (size_t)128 * NN;

  const size_t lds1 = (size_t)(640 * 136 + 3 * 128 * 136) * 2; // 278528 B
  k1_fused_in<<<NN / 128, 256, lds1, stream>>>(pair, ln_in_w, ln_in_b, p_in_w,
                                               g_in_w, g_out_w, leftT, rightT,
                                               gateT);

  const size_t lds3 = (size_t)(2 * K3_BUFSZ) * 2; // 40960 B (double buffer)
  k3_triangle<<<128 * 16, 256, lds3, stream>>>(leftT, rightT, outT);

  const size_t lds4 = (size_t)256 * 132 * 4 + (size_t)(256 * 136 + 128 * 136) * 2; // 239616 B
  k4_out<<<(N_SEQ / 16) * (N_SEQ / 16), 256, lds4, stream>>>(
      outT, gateT, ln_out_w, ln_out_b, p_out_w, (float*)d_out);
}
